// QuestionGuidedGraphAttention_38482906972435
// MI455X (gfx1250) — matrix operand order repaired and checked by reading, not compiled, not run on hardware
//
#include <hip/hip_runtime.h>
#include <hip/hip_bf16.h>

typedef __attribute__((ext_vector_type(16))) __bf16 v16bf;
typedef __attribute__((ext_vector_type(8)))  float  v8f;
typedef __attribute__((ext_vector_type(4)))  float  f32x4;
typedef __attribute__((ext_vector_type(4)))  int    i32x4;

#if defined(__AMDGCN__) && __has_builtin(__builtin_amdgcn_global_load_async_to_lds_b128) && __has_builtin(__builtin_amdgcn_s_wait_asynccnt)
#define ASYNC_LDS 1
typedef __attribute__((address_space(1))) i32x4 as1_i32x4;
typedef __attribute__((address_space(3))) i32x4 as3_i32x4;
#else
#define ASYNC_LDS 0
#endif

// ---------------------------------------------------------------------------
// Kernel 1: question projections qn = question @ W_nq + b_nq (and qe).
// 2 * 64 * 512 = 65536 outputs, one thread each. Tiny, cache-resident.
// ---------------------------------------------------------------------------
__global__ void __launch_bounds__(256)
qproj_kernel(const float* __restrict__ question,
             const float* __restrict__ Wnq, const float* __restrict__ bnq,
             const float* __restrict__ Weq, const float* __restrict__ beq,
             float* __restrict__ qn, float* __restrict__ qe) {
    int idx   = blockIdx.x * 256 + threadIdx.x;   // 0..65535
    int p     = idx & 511;
    int b     = (idx >> 9) & 63;
    int which = idx >> 15;
    const float* W    = which ? Weq : Wnq;
    const float* bias = which ? beq : bnq;
    const float* qr   = question + (size_t)b * 768;
    float acc = 0.f;
    #pragma unroll 4
    for (int k = 0; k < 768; ++k)
        acc += qr[k] * W[(size_t)k * 512 + p];
    float* out = which ? qe : qn;
    out[(size_t)b * 512 + p] = acc + bias[p];
}

// ---------------------------------------------------------------------------
// Kernel 2: pack W_n / W_e (f32 [K=512, P=512], row-major) into bf16 WMMA
// B-fragments. Per fragment (nt, kt): lane<16 holds K=kt*32+0..15 of column
// P=nt*16+lane; lane>=16 holds K=kt*32+16..31 of column P=nt*16+(lane-16).
// Packed layout: matrix_base + ((nt*16 + kt)*32 + lane)*32 bytes, so one
// N-panel (nt) is a contiguous 16 KB blob -> straight async-copy into LDS.
// ---------------------------------------------------------------------------
__global__ void __launch_bounds__(256)
pack_w_kernel(const float* __restrict__ Wn, const float* __restrict__ We,
              char* __restrict__ packed) {
    int idx   = blockIdx.x * 256 + threadIdx.x;   // 0..32767
    int which = idx >> 14;
    int e     = idx & 16383;                      // nt*512 + kt*32 + lane
    int lane  = e & 31;
    int kt    = (e >> 5) & 15;
    int nt    = e >> 9;
    const float* W = which ? We : Wn;
    int p     = nt * 16 + (lane & 15);
    int kbase = kt * 32 + ((lane >> 4) << 4);
    v16bf v;
    #pragma unroll
    for (int r = 0; r < 16; ++r)
        v[r] = (__bf16)W[(size_t)(kbase + r) * 512 + p];
    *(v16bf*)(packed + (size_t)which * 524288 + (size_t)e * 32) = v;
}

// ---------------------------------------------------------------------------
// Kernel 3: fused GEMM + tanh + dot(w_v) -> per-row score.
// Block = 256 threads = 8 waves; each wave owns a 16-row tile (block: 128
// rows). A-strip (16 rows x 512 K) held in registers as 16 bf16 fragments.
// 32 N-panels, double-buffered: async-copy panel nt+1 global->LDS (no VGPR
// round trip, ASYNCcnt-tracked) while the 16 chained
// v_wmma_f32_16x16x32_bf16 of panel nt execute.
// C layout (ISA 7.12.2): vgpr j, lane l: M = j + (l>=16 ? 8 : 0), N = l&15.
// ---------------------------------------------------------------------------
__global__ void __launch_bounds__(256)
score_kernel(const float* __restrict__ A,        // [Mtot, 512] f32
             const char*  __restrict__ packedW,  // bf16 fragments
             const float* __restrict__ bias,     // [512]
             const float* __restrict__ qtab,     // [64, 512] precomputed q-proj
             const float* __restrict__ wv,       // [512]
             const float* __restrict__ bvp,      // scalar bias
             const int*   __restrict__ gid,      // [Mtot]
             float*       __restrict__ sout) {   // [Mtot] scores
    __shared__ __align__(32) char ldsB[2][16384];

    const int tid  = threadIdx.x;
    const int wave = tid >> 5;
    const int lane = tid & 31;
    const int half = lane >> 4;
    const int lr   = lane & 15;
    const int r0   = (blockIdx.x * 8 + wave) * 16;

    // Issue copy of one 16 KB B-panel into ldsB[buf]; 64 contiguous B/thread.
    auto issue_copy = [&](int buf, int nt) {
#if ASYNC_LDS
        as1_i32x4* g = (as1_i32x4*)(packedW + (size_t)nt * 16384 + tid * 64);
        as3_i32x4* l = (as3_i32x4*)(&ldsB[buf][tid * 64]);
        __builtin_amdgcn_global_load_async_to_lds_b128(g, l, 0,  0);
        __builtin_amdgcn_global_load_async_to_lds_b128(g, l, 16, 0);
        __builtin_amdgcn_global_load_async_to_lds_b128(g, l, 32, 0);
        __builtin_amdgcn_global_load_async_to_lds_b128(g, l, 48, 0);
#else
        const f32x4* s = (const f32x4*)(packedW + (size_t)nt * 16384 + tid * 64);
        f32x4 t0 = s[0], t1 = s[1], t2 = s[2], t3 = s[3];
        f32x4* d = (f32x4*)(&ldsB[buf][tid * 64]);
        d[0] = t0; d[1] = t1; d[2] = t2; d[3] = t3;
#endif
    };

    // --- Load the 16x512 A-strip (streamed once -> nontemporal), converting
    // f32 -> bf16 into WMMA A layout. lane<16: row M=lr, K runs
    // c..c+15; lane>=16: c+16..c+31.
    v16bf afr[16];
    const float* arow = A + (size_t)(r0 + lr) * 512;
    #pragma unroll
    for (int kt = 0; kt < 16; ++kt) {
        const int c1 = kt * 32 + half * 16;
        f32x4 f0 = __builtin_nontemporal_load((const f32x4*)(arow + c1));
        f32x4 f1 = __builtin_nontemporal_load((const f32x4*)(arow + c1 + 4));
        f32x4 f2 = __builtin_nontemporal_load((const f32x4*)(arow + c1 + 8));
        f32x4 f3 = __builtin_nontemporal_load((const f32x4*)(arow + c1 + 12));
        v16bf a;
        a[0]  = (__bf16)f0[0]; a[1]  = (__bf16)f0[1]; a[2]  = (__bf16)f0[2]; a[3]  = (__bf16)f0[3];
        a[4]  = (__bf16)f1[0]; a[5]  = (__bf16)f1[1]; a[6]  = (__bf16)f1[2]; a[7]  = (__bf16)f1[3];
        a[8]  = (__bf16)f2[0]; a[9]  = (__bf16)f2[1]; a[10] = (__bf16)f2[2]; a[11] = (__bf16)f2[3];
        a[12] = (__bf16)f3[0]; a[13] = (__bf16)f3[1]; a[14] = (__bf16)f3[2]; a[15] = (__bf16)f3[3];
        afr[kt] = a;
    }

    // Graph ids for the 8 rows this lane's C fragments belong to.
    int g8[8];
    #pragma unroll
    for (int j = 0; j < 8; ++j) g8[j] = gid[r0 + half * 8 + j];

    float acc[8] = {0.f, 0.f, 0.f, 0.f, 0.f, 0.f, 0.f, 0.f};

    issue_copy(0, 0);   // prologue: panel 0 in flight

    for (int nt = 0; nt < 32; ++nt) {
        const int buf = nt & 1;
#if ASYNC_LDS
        __builtin_amdgcn_s_wait_asynccnt(0);   // this thread's panel landed
#endif
        __syncthreads();                        // all landed; prev readers done
        if (nt < 31) issue_copy(buf ^ 1, nt + 1);   // overlaps with WMMAs below

        v8f c = {};
        #pragma unroll
        for (int kt = 0; kt < 16; ++kt) {
            v16bf b = *(const v16bf*)(&ldsB[buf][(size_t)(kt * 32 + lane) * 32]);
            c = __builtin_amdgcn_wmma_f32_16x16x32_bf16(
                    false, afr[kt], false, b, (short)0, c, false, false);
        }

        // Epilogue: x = c + bias[p] + q[gid, p]; tanh(x) * w_v[p] accumulate.
        const int   col = nt * 16 + lr;
        const float w   = wv[col];
        const float bb  = bias[col];
        #pragma unroll
        for (int j = 0; j < 8; ++j) {
            float x  = c[j] + bb + qtab[(size_t)g8[j] * 512 + col];
            float e2 = __expf(2.f * x);                              // v_exp_f32
            float t  = 1.f - 2.f * __builtin_amdgcn_rcpf(e2 + 1.f);  // tanh
            acc[j] += t * w;
        }
    }

    // Cross-lane reduce: rows j live in lanes 0-15 (M=j) / 16-31 (M=j+8);
    // xor masks 1/2/4/8 reduce within each 16-lane half.
    const float bv = bvp[0];
    #pragma unroll
    for (int j = 0; j < 8; ++j) {
        float v = acc[j];
        v += __shfl_xor(v, 1, 32);
        v += __shfl_xor(v, 2, 32);
        v += __shfl_xor(v, 4, 32);
        v += __shfl_xor(v, 8, 32);
        if (lr == 0) sout[r0 + half * 8 + j] = v + bv;
    }
}

// ---------------------------------------------------------------------------
// Kernel 4: deterministic segment max / sum-exp. One block per (array, g):
// binary search sorted gid for [lo,hi), tree-reduce. No float atomics ->
// bitwise deterministic across replays.
// ---------------------------------------------------------------------------
__device__ __forceinline__ int lower_bound_dev(const int* a, int n, int key) {
    int lo = 0, hi = n;
    while (lo < hi) { int mid = (lo + hi) >> 1; if (a[mid] < key) lo = mid + 1; else hi = mid; }
    return lo;
}

__global__ void __launch_bounds__(256)
seg_stats_kernel(const float* __restrict__ s_n, const int* __restrict__ gid_n, int n_n,
                 const float* __restrict__ s_e, const int* __restrict__ gid_e, int n_e,
                 float* __restrict__ m_out, float* __restrict__ z_out) {
    int blk = blockIdx.x, which = blk >> 6, g = blk & 63;
    const float* s   = which ? s_e   : s_n;
    const int*   gid = which ? gid_e : gid_n;
    int n = which ? n_e : n_n;

    __shared__ int sh_lo, sh_hi;
    __shared__ float red[256];
    __shared__ float sh_m;
    int tid = threadIdx.x;
    if (tid == 0) {
        sh_lo = lower_bound_dev(gid, n, g);
        sh_hi = lower_bound_dev(gid, n, g + 1);
    }
    __syncthreads();
    int lo = sh_lo, hi = sh_hi;

    float lm = -3.402823466e38f;
    for (int i = lo + tid; i < hi; i += 256) lm = fmaxf(lm, s[i]);
    red[tid] = lm; __syncthreads();
    for (int off = 128; off; off >>= 1) {
        if (tid < off) red[tid] = fmaxf(red[tid], red[tid + off]);
        __syncthreads();
    }
    if (tid == 0) sh_m = red[0];
    __syncthreads();
    float m = sh_m;

    float lz = 0.f;
    for (int i = lo + tid; i < hi; i += 256) lz += __expf(s[i] - m);
    red[tid] = lz; __syncthreads();
    for (int off = 128; off; off >>= 1) {
        if (tid < off) red[tid] += red[tid + off];
        __syncthreads();
    }
    if (tid == 0) { m_out[which * 64 + g] = m; z_out[which * 64 + g] = red[0]; }
}

// ---------------------------------------------------------------------------
// Kernel 5: att = exp(s - m[gid]) / z[gid], written to concatenated d_out.
// ---------------------------------------------------------------------------
__global__ void __launch_bounds__(256)
finalize_kernel(const float* __restrict__ s_n, const int* __restrict__ gid_n, int n_n,
                const float* __restrict__ s_e, const int* __restrict__ gid_e, int n_e,
                const float* __restrict__ m, const float* __restrict__ z,
                float* __restrict__ out) {
    int i = blockIdx.x * 256 + threadIdx.x;
    if (i < n_n) {
        int g = gid_n[i];
        out[i] = __expf(s_n[i] - m[g]) / z[g];
    } else {
        int j = i - n_n;
        if (j < n_e) {
            int g = gid_e[j];
            out[n_n + j] = __expf(s_e[j] - m[64 + g]) / z[64 + g];
        }
    }
}

// ---------------------------------------------------------------------------
extern "C" void kernel_launch(void* const* d_in, const int* in_sizes, int n_in,
                              void* d_out, int out_size, void* d_ws, size_t ws_size,
                              hipStream_t stream) {
    const float* question = (const float*)d_in[0];
    const float* nodes    = (const float*)d_in[1];
    const float* edges    = (const float*)d_in[2];
    const float* W_nq     = (const float*)d_in[3];
    const float* b_nq     = (const float*)d_in[4];
    const float* W_n      = (const float*)d_in[5];
    const float* b_n      = (const float*)d_in[6];
    const float* w_nv     = (const float*)d_in[7];
    const float* b_nv     = (const float*)d_in[8];
    const float* W_eq     = (const float*)d_in[9];
    const float* b_eq     = (const float*)d_in[10];
    const float* W_e      = (const float*)d_in[11];
    const float* b_e      = (const float*)d_in[12];
    const float* w_ev     = (const float*)d_in[13];
    const float* b_ev     = (const float*)d_in[14];
    const int* node_gid   = (const int*)d_in[15];
    const int* edge_gid   = (const int*)d_in[16];

    const int TN = 131072, TE = 262144;

    // Workspace layout (bytes):
    char*  ws     = (char*)d_ws;
    float* qn     = (float*)(ws + 0);          // 64*512*4   = 131072
    float* qe     = (float*)(ws + 131072);     // 131072
    char*  packed = ws + 262144;               // 2 * 512 KB = 1048576
    float* s_n    = (float*)(ws + 1310720);    // TN*4       = 524288
    float* s_e    = (float*)(ws + 1835008);    // TE*4       = 1048576
    float* mbuf   = (float*)(ws + 2883584);    // 128 floats
    float* zbuf   = (float*)(ws + 2884096);    // 128 floats
    float* out    = (float*)d_out;

    qproj_kernel<<<256, 256, 0, stream>>>(question, W_nq, b_nq, W_eq, b_eq, qn, qe);
    pack_w_kernel<<<128, 256, 0, stream>>>(W_n, W_e, packed);
    score_kernel<<<TN / 128, 256, 0, stream>>>(nodes, packed,          b_n, qn, w_nv, b_nv, node_gid, s_n);
    score_kernel<<<TE / 128, 256, 0, stream>>>(edges, packed + 524288, b_e, qe, w_ev, b_ev, edge_gid, s_e);
    seg_stats_kernel<<<128, 256, 0, stream>>>(s_n, node_gid, TN, s_e, edge_gid, TE, mbuf, zbuf);
    finalize_kernel<<<(TN + TE) / 256, 256, 0, stream>>>(s_n, node_gid, TN, s_e, edge_gid, TE, mbuf, zbuf, out);
}
